// Block_30313879175568
// MI455X (gfx1250) — matrix operand order repaired and checked by reading, not compiled, not run on hardware
//
#include <hip/hip_runtime.h>
#include <cstdint>

// ---------------- problem constants ----------------
constexpr int B_   = 4;
constexpr int L_   = 2048;
constexpr int DIM_ = 1024;
constexpr int H_   = 16;
constexpr int HID_ = 4096;
constexpr int HD_  = 64;
constexpr int R_   = B_ * L_;          // 8192 rows of activations

typedef __attribute__((ext_vector_type(16))) __bf16 v16bf;
typedef __attribute__((ext_vector_type(8)))  float  v8f;
typedef __attribute__((ext_vector_type(4)))  unsigned int u32x4;
typedef __attribute__((ext_vector_type(8)))  int  i32x8;
typedef __attribute__((ext_vector_type(4)))  int  i32x4;

#if __has_builtin(__builtin_amdgcn_tensor_load_to_lds)
#define HAVE_TDM 1
#else
#define HAVE_TDM 0
#endif

union Frag {
  v16bf v;
  uint4 q[2];
  unsigned short s[16];
};

static __device__ __forceinline__ unsigned short f2bf(float f) {
  unsigned u = __builtin_bit_cast(unsigned, f);
  u += 0x7fffu + ((u >> 16) & 1u);          // round-to-nearest-even
  return (unsigned short)(u >> 16);
}

#if HAVE_TDM
// Issue a TDM load of a 128x32 bf16 tile (row stride K elements) into LDS,
// with 16B of LDS padding after every 64B row (matches ushort[...][40] layout).
static __device__ __forceinline__ void tdm_load_tile_a(const unsigned short* gptr,
                                                       unsigned lds_off, int K) {
  unsigned long long ga = (unsigned long long)(uintptr_t)gptr;
  u32x4 g0;
  g0.x = 1u;                                         // count=1, user mode
  g0.y = lds_off;                                    // lds_addr (bytes)
  g0.z = (unsigned)(ga & 0xFFFFFFFFull);             // global_addr[31:0]
  g0.w = (unsigned)((ga >> 32) & 0x1FFFFFFull) | (2u << 30);  // addr[56:32] | type=2
  i32x8 g1;
  g1[0] = (1 << 16) | (1 << 20) | (3 << 22) | (3 << 25); // data_size=2B, pad_en, ivl=16DW, amt=4DW
  g1[1] = (K & 0xFFFF) << 16;                        // tensor_dim0[15:0] = K
  g1[2] = (128 << 16);                               // tensor_dim1[15:0] = 128 rows
  g1[3] = (32 << 16);                                // tile_dim0 = 32
  g1[4] = 128;                                       // tile_dim1 = 128
  g1[5] = K;                                         // tensor_dim0_stride = K
  g1[6] = 0;
  g1[7] = 0;
  i32x4 z4 = {0, 0, 0, 0};
  i32x8 z8 = {0, 0, 0, 0, 0, 0, 0, 0};
  __builtin_amdgcn_tensor_load_to_lds(g0, g1, z4, z4, z8, 0);
}
#endif

// ---------------- fp32 -> bf16 cast (weights) ----------------
__global__ void cast_bf16_kernel(const float* __restrict__ in,
                                 unsigned short* __restrict__ out, int n) {
  int i = blockIdx.x * blockDim.x + threadIdx.x;
  int stride = gridDim.x * blockDim.x;
  for (; i < n; i += stride) out[i] = f2bf(in[i]);
}

// ---------------- RMSNorm (row of 1024) -> bf16 ----------------
__global__ __launch_bounds__(256) void rmsnorm_kernel(const float* __restrict__ x,
                                                      const float* __restrict__ scale,
                                                      unsigned short* __restrict__ out,
                                                      float eps) {
  __shared__ float red[256];
  const size_t row = blockIdx.x;
  const float* xr = x + row * DIM_;
  int t = threadIdx.x;
  float v[4];
  float ss = 0.f;
#pragma unroll
  for (int k = 0; k < 4; ++k) { v[k] = xr[t + 256 * k]; ss += v[k] * v[k]; }
  red[t] = ss; __syncthreads();
  for (int s = 128; s > 0; s >>= 1) { if (t < s) red[t] += red[t + s]; __syncthreads(); }
  float inv = rsqrtf(red[0] / (float)DIM_ + eps);
#pragma unroll
  for (int k = 0; k < 4; ++k) {
    int i = t + 256 * k;
    out[row * DIM_ + i] = f2bf(v[k] * inv * scale[i]);
  }
}

// ---------------- tiled WMMA GEMM: C = A(bf16) @ B(bf16) + bias (+resid) ----------------
// BM=128, BN=128, BK=32; 8 waves, each computes 64x32 (4x2 tiles of 16x16).
// Double-buffered: A tile staged by TDM (async DMA -> LDS), B tile staged with
// a vector-load transpose; tile k+1 staged while tile k is consumed by WMMA.
template <bool RES>
__global__ __launch_bounds__(256) void gemm_bf16(const unsigned short* __restrict__ A,
                                                 const unsigned short* __restrict__ Bm,
                                                 const float* __restrict__ bias,
                                                 const float* __restrict__ resid,
                                                 float* __restrict__ C,
                                                 int M, int N, int K) {
  __shared__ unsigned short As[2][129][40];  // +1 guard row absorbs trailing DMA pad
  __shared__ unsigned short Bs[2][128][40];  // transposed: Bs[buf][n][k]
  const int tid  = threadIdx.x;
  const int lane = tid & 31;
  const int wave = tid >> 5;
  const int wm   = wave >> 2, wn = wave & 3;
  const int lm   = lane & 15;
  const int hsel = lane >> 4;
  const int blockM = blockIdx.y * 128;
  const int blockN = blockIdx.x * 128;

  v8f acc[4][2] = {};

  auto issueA = [&](int kb, int buf) {
#if HAVE_TDM
    if (wave == 0) {
      unsigned lds_off = (unsigned)(uintptr_t)&As[buf][0][0];
      tdm_load_tile_a(A + (size_t)blockM * K + kb, lds_off, K);
    }
#else
#pragma unroll
    for (int p = 0; p < 2; ++p) {
      int u = tid + p * 256;                 // 512 uint4 units
      int r = u >> 2, kc = (u & 3) * 8;
      *(uint4*)&As[buf][r][kc] = *(const uint4*)&A[(size_t)(blockM + r) * K + kb + kc];
    }
#endif
  };
  auto stageB = [&](int kb, int buf) {
#pragma unroll
    for (int p = 0; p < 2; ++p) {
      int u = tid + p * 256;
      int kr = u >> 4, n8 = (u & 15) * 8;
      uint4 d = *(const uint4*)&Bm[(size_t)(kb + kr) * N + blockN + n8];
      const unsigned short* ds = (const unsigned short*)&d;
#pragma unroll
      for (int i = 0; i < 8; ++i) Bs[buf][n8 + i][kr] = ds[i];
    }
  };

  // prologue: stage tile 0
  issueA(0, 0);
  stageB(0, 0);
#if HAVE_TDM
  if (wave == 0) __builtin_amdgcn_s_wait_tensorcnt(0);
#endif
  __syncthreads();

  int cur = 0;
  for (int kb = 0; kb < K; kb += 32) {
    int nxt = cur ^ 1;
    bool more = (kb + 32) < K;
    if (more) {
      issueA(kb + 32, nxt);                  // async DMA overlaps WMMA below
      stageB(kb + 32, nxt);
      __builtin_prefetch(&Bm[(size_t)(kb + 64) * N + blockN], 0, 0);
    }

    Frag a[4], b[2];
#pragma unroll
    for (int mi = 0; mi < 4; ++mi) {       // 16-bit A layout: lane m, K chunks 0/8 and 16/24
      int m  = wm * 64 + mi * 16 + lm;
      int k1 = hsel * 8;
      a[mi].q[0] = *(const uint4*)&As[cur][m][k1];
      a[mi].q[1] = *(const uint4*)&As[cur][m][16 + k1];
    }
#pragma unroll
    for (int ni = 0; ni < 2; ++ni) {       // 16-bit B layout: lane n, K chunks 0/8 and 16/24
      int n  = wn * 32 + ni * 16 + lm;
      int kk = hsel * 8;
      b[ni].q[0] = *(const uint4*)&Bs[cur][n][kk];
      b[ni].q[1] = *(const uint4*)&Bs[cur][n][16 + kk];
    }
#pragma unroll
    for (int mi = 0; mi < 4; ++mi)
#pragma unroll
      for (int ni = 0; ni < 2; ++ni)
        acc[mi][ni] = __builtin_amdgcn_wmma_f32_16x16x32_bf16(
            false, a[mi].v, false, b[ni].v, (short)0, acc[mi][ni], false, false);

#if HAVE_TDM
    if (wave == 0 && more) __builtin_amdgcn_s_wait_tensorcnt(0);
#endif
    __syncthreads();
    cur = nxt;
  }

  // epilogue: bias (+ residual), fp32 out.  C layout: VGPR r -> row r / r+8
#pragma unroll
  for (int mi = 0; mi < 4; ++mi) {
#pragma unroll
    for (int ni = 0; ni < 2; ++ni) {
      int coln = blockN + wn * 32 + ni * 16 + lm;
      float bv = bias[coln];
#pragma unroll
      for (int r = 0; r < 8; ++r) {
        int rowm = blockM + wm * 64 + mi * 16 + hsel * 8 + r;
        size_t idx = (size_t)rowm * N + coln;
        float val = acc[mi][ni][r] + bv;
        if (RES) val += resid[idx];
        C[idx] = val;
      }
    }
  }
}

// ---------------- RoPE + head transpose + bf16 cast ----------------
__global__ __launch_bounds__(32) void rope_kernel(const float* __restrict__ qkv,
                                                  unsigned short* __restrict__ Qb,
                                                  unsigned short* __restrict__ Kb,
                                                  unsigned short* __restrict__ Vb) {
  const int i = threadIdx.x;                 // 0..31 (half of HD)
  const int pos = blockIdx.x, h = blockIdx.y, b = blockIdx.z;
  const size_t row = (size_t)b * L_ + pos;
  const float* src = qkv + row * (3 * DIM_);
  float invf = __powf(10000.f, -(float)(2 * i) / (float)HD_);
  float th = (float)pos * invf;
  float sn, cs;
  __sincosf(th, &sn, &cs);
  int o = h * HD_;
  float q1 = src[o + i],            q2 = src[o + 32 + i];
  float k1 = src[DIM_ + o + i],     k2 = src[DIM_ + o + 32 + i];
  float v1 = src[2 * DIM_ + o + i], v2 = src[2 * DIM_ + o + 32 + i];
  size_t dst = (((size_t)(b * H_ + h)) * L_ + pos) * HD_;
  Qb[dst + i]      = f2bf(q1 * cs - q2 * sn);
  Qb[dst + 32 + i] = f2bf(q1 * sn + q2 * cs);
  Kb[dst + i]      = f2bf(k1 * cs - k2 * sn);
  Kb[dst + 32 + i] = f2bf(k1 * sn + k2 * cs);
  Vb[dst + i]      = f2bf(v1);
  Vb[dst + 32 + i] = f2bf(v2);
}

// ---------------- flash-style causal attention (one wave / 16 q-rows) ----------------
__global__ __launch_bounds__(32) void attn_kernel(const unsigned short* __restrict__ Q,
                                                  const unsigned short* __restrict__ Kt,
                                                  const unsigned short* __restrict__ V,
                                                  unsigned short* __restrict__ O) {
  __shared__ unsigned short Plds[16][40];
  const int lane = threadIdx.x;
  const int lm = lane & 15, hsel = lane >> 4;
  const int qt = blockIdx.x, h = blockIdx.y, b = blockIdx.z;
  const int qbase = qt * 16;
  const size_t headbase = ((size_t)(b * H_ + h)) * L_;

  // Q fragments (A-layout), K-dim halves of HD=64, kept resident
  Frag aqlo, aqhi;
  {
    const unsigned short* qp = Q + (headbase + qbase + lm) * HD_;
    int k1 = hsel * 8;
    aqlo.q[0] = *(const uint4*)(qp + k1);
    aqlo.q[1] = *(const uint4*)(qp + 16 + k1);
    aqhi.q[0] = *(const uint4*)(qp + 32 + k1);
    aqhi.q[1] = *(const uint4*)(qp + 48 + k1);
  }

  float mrun[8], lrun[8];
#pragma unroll
  for (int r = 0; r < 8; ++r) { mrun[r] = -3.0e38f; lrun[r] = 0.f; }
  v8f oacc[4] = {};
  const float scale = 0.125f;  // 1/sqrt(64)

  for (int kv0 = 0; kv0 < qbase + 16; kv0 += 32) {
    v8f sc[2];
#pragma unroll
    for (int nt = 0; nt < 2; ++nt) {       // scores: Q[16x64] @ K^T -> 16x16 per nt
      const unsigned short* kp = Kt + (headbase + kv0 + nt * 16 + lm) * HD_;
      int kk = hsel * 8;
      Frag blo, bhi;
      blo.q[0] = *(const uint4*)(kp + kk);
      blo.q[1] = *(const uint4*)(kp + 16 + kk);
      bhi.q[0] = *(const uint4*)(kp + 32 + kk);
      bhi.q[1] = *(const uint4*)(kp + 48 + kk);
      v8f c0 = {};
      c0 = __builtin_amdgcn_wmma_f32_16x16x32_bf16(false, aqlo.v, false, blo.v, (short)0, c0, false, false);
      c0 = __builtin_amdgcn_wmma_f32_16x16x32_bf16(false, aqhi.v, false, bhi.v, (short)0, c0, false, false);
      sc[nt] = c0;
    }
    // online softmax; row r lives in VGPR r across the 16 lanes of each half-wave
#pragma unroll
    for (int r = 0; r < 8; ++r) {
      int row = qbase + hsel * 8 + r;
      float s0 = sc[0][r] * scale + ((kv0 + lm)      <= row ? 0.f : -1.0e9f);
      float s1 = sc[1][r] * scale + ((kv0 + 16 + lm) <= row ? 0.f : -1.0e9f);
      float tm = fmaxf(s0, s1);
#pragma unroll
      for (int off = 8; off > 0; off >>= 1) tm = fmaxf(tm, __shfl_xor(tm, off, 32));
      float mnew  = fmaxf(mrun[r], tm);
      float alpha = __expf(mrun[r] - mnew);
      float p0 = __expf(s0 - mnew);
      float p1 = __expf(s1 - mnew);
      float rs = p0 + p1;
#pragma unroll
      for (int off = 8; off > 0; off >>= 1) rs += __shfl_xor(rs, off, 32);
      lrun[r] = lrun[r] * alpha + rs;
      mrun[r] = mnew;
#pragma unroll
      for (int ch = 0; ch < 4; ++ch) oacc[ch][r] *= alpha;
      // C-layout -> LDS (row-major P tile) for A-layout reload
      Plds[hsel * 8 + r][lm]      = f2bf(p0);
      Plds[hsel * 8 + r][16 + lm] = f2bf(p1);
    }
    __syncthreads();
    Frag ap;
    {
      int k1 = hsel * 8;
      ap.q[0] = *(const uint4*)&Plds[lm][k1];
      ap.q[1] = *(const uint4*)&Plds[lm][16 + k1];
    }
#pragma unroll
    for (int ch = 0; ch < 4; ++ch) {       // O[16x64] += P[16x32] @ V[32x64]
      Frag bv;
#pragma unroll
      for (int j = 0; j < 16; ++j)
        bv.s[j] = V[(headbase + kv0 + hsel * 8 + (j & 7) + (j >> 3) * 16) * HD_ + ch * 16 + lm];
      oacc[ch] = __builtin_amdgcn_wmma_f32_16x16x32_bf16(
          false, ap.v, false, bv.v, (short)0, oacc[ch], false, false);
    }
    __syncthreads();
  }

  // normalize, cast, write back in [B*L, DIM] layout (heads re-interleaved)
#pragma unroll
  for (int ch = 0; ch < 4; ++ch)
#pragma unroll
    for (int r = 0; r < 8; ++r) {
      int row = qbase + hsel * 8 + r;
      O[((size_t)b * L_ + row) * DIM_ + h * HD_ + ch * 16 + lm] =
          f2bf(oacc[ch][r] / lrun[r]);
    }
}

// ---------------- GELU (exact) + LayerNorm (row of 4096) -> bf16 ----------------
__global__ __launch_bounds__(256) void gelu_ln_kernel(const float* __restrict__ g,
                                                      const float* __restrict__ gamma,
                                                      const float* __restrict__ beta,
                                                      unsigned short* __restrict__ out) {
  __shared__ float red[256], red2[256];
  const size_t row = blockIdx.x;
  const float* gr = g + row * HID_;
  int t = threadIdx.x;
  float v[16];
  float s = 0.f, s2 = 0.f;
#pragma unroll
  for (int k = 0; k < 16; ++k) {
    float x = gr[t + 256 * k];
    float gl = 0.5f * x * (1.f + erff(x * 0.70710678118f));
    v[k] = gl; s += gl; s2 += gl * gl;
  }
  red[t] = s; red2[t] = s2; __syncthreads();
  for (int st = 128; st > 0; st >>= 1) {
    if (t < st) { red[t] += red[t + st]; red2[t] += red2[t + st]; }
    __syncthreads();
  }
  float mu  = red[0] / (float)HID_;
  float var = red2[0] / (float)HID_ - mu * mu;
  float inv = rsqrtf(var + 1e-5f);
#pragma unroll
  for (int k = 0; k < 16; ++k) {
    int i = t + 256 * k;
    out[row * HID_ + i] = f2bf((v[k] - mu) * inv * gamma[i] + beta[i]);
  }
}

// ---------------- launcher ----------------
extern "C" void kernel_launch(void* const* d_in, const int* in_sizes, int n_in,
                              void* d_out, int out_size, void* d_ws, size_t ws_size,
                              hipStream_t stream) {
  (void)in_sizes; (void)n_in; (void)out_size; (void)ws_size;
  const float* x       = (const float*)d_in[0];
  /* d_in[1] mask: causal mask computed analytically in-kernel */
  const float* attn_sc = (const float*)d_in[2];
  const float* wqkv_w  = (const float*)d_in[3];
  const float* wqkv_b  = (const float*)d_in[4];
  const float* out_w   = (const float*)d_in[5];
  const float* out_b   = (const float*)d_in[6];
  const float* ffn_sc  = (const float*)d_in[7];
  const float* lin1_w  = (const float*)d_in[8];
  const float* lin1_b  = (const float*)d_in[9];
  const float* ln_g    = (const float*)d_in[10];
  const float* ln_b    = (const float*)d_in[11];
  const float* lin2_w  = (const float*)d_in[12];
  const float* lin2_b  = (const float*)d_in[13];
  float* out = (float*)d_out;

  char* wsp = (char*)d_ws;
  auto alloc = [&](size_t bytes) -> char* {
    char* p = wsp; wsp += (bytes + 255) & ~(size_t)255; return p;
  };
  unsigned short* wqkv_bf = (unsigned short*)alloc((size_t)DIM_ * 3 * DIM_ * 2);
  unsigned short* wout_bf = (unsigned short*)alloc((size_t)DIM_ * DIM_ * 2);
  unsigned short* wl1_bf  = (unsigned short*)alloc((size_t)DIM_ * HID_ * 2);
  unsigned short* wl2_bf  = (unsigned short*)alloc((size_t)HID_ * DIM_ * 2);
  unsigned short* xn_bf   = (unsigned short*)alloc((size_t)R_ * DIM_ * 2);
  float*          qkv_f   = (float*)alloc((size_t)R_ * 3 * DIM_ * 4);
  unsigned short* q_bf    = (unsigned short*)alloc((size_t)R_ * DIM_ * 2);
  unsigned short* k_bf    = (unsigned short*)alloc((size_t)R_ * DIM_ * 2);
  unsigned short* v_bf    = (unsigned short*)alloc((size_t)R_ * DIM_ * 2);
  unsigned short* o_bf    = (unsigned short*)alloc((size_t)R_ * DIM_ * 2);
  float*          h_f     = (float*)alloc((size_t)R_ * DIM_ * 4);
  unsigned short* f_bf    = (unsigned short*)alloc((size_t)R_ * DIM_ * 2);
  float*          g_f     = (float*)alloc((size_t)R_ * HID_ * 4);
  unsigned short* f2_bf   = (unsigned short*)alloc((size_t)R_ * HID_ * 2);

  // weights -> bf16
  cast_bf16_kernel<<<512, 256, 0, stream>>>(wqkv_w, wqkv_bf, DIM_ * 3 * DIM_);
  cast_bf16_kernel<<<512, 256, 0, stream>>>(out_w,  wout_bf, DIM_ * DIM_);
  cast_bf16_kernel<<<512, 256, 0, stream>>>(lin1_w, wl1_bf,  DIM_ * HID_);
  cast_bf16_kernel<<<512, 256, 0, stream>>>(lin2_w, wl2_bf,  HID_ * DIM_);

  // attention half
  rmsnorm_kernel<<<R_, 256, 0, stream>>>(x, attn_sc, xn_bf, 1e-6f);
  gemm_bf16<false><<<dim3(3 * DIM_ / 128, R_ / 128), 256, 0, stream>>>(
      xn_bf, wqkv_bf, wqkv_b, nullptr, qkv_f, R_, 3 * DIM_, DIM_);
  rope_kernel<<<dim3(L_, H_, B_), 32, 0, stream>>>(qkv_f, q_bf, k_bf, v_bf);
  attn_kernel<<<dim3(L_ / 16, H_, B_), 32, 0, stream>>>(q_bf, k_bf, v_bf, o_bf);
  gemm_bf16<true><<<dim3(DIM_ / 128, R_ / 128), 256, 0, stream>>>(
      o_bf, wout_bf, out_b, x, h_f, R_, DIM_, DIM_);

  // FFN half
  rmsnorm_kernel<<<R_, 256, 0, stream>>>(h_f, ffn_sc, f_bf, 1e-6f);
  gemm_bf16<false><<<dim3(HID_ / 128, R_ / 128), 256, 0, stream>>>(
      f_bf, wl1_bf, lin1_b, nullptr, g_f, R_, HID_, DIM_);
  gelu_ln_kernel<<<R_, 256, 0, stream>>>(g_f, ln_g, ln_b, f2_bf);
  gemm_bf16<true><<<dim3(DIM_ / 128, R_ / 128), 256, 0, stream>>>(
      f2_bf, wl2_bf, lin2_b, h_f, out, R_, DIM_, HID_);
}
